// SparseTransformer_87436944212086
// MI455X (gfx1250) — matrix operand order repaired and checked by reading, not compiled, not run on hardware
//
#include <hip/hip_runtime.h>
#include <hip/hip_bf16.h>

// ---------------------------------------------------------------------------
// Model constants (from reference)
// ---------------------------------------------------------------------------
#define BATCH    64
#define TLEN     128
#define DMOT     256
#define DM       1024
#define NLAYERS  8
#define NHEADS   16
#define DFF      4096
#define DHEAD    64
#define CTXN     10
#define NTOK     (BATCH * TLEN)     // 8192
#define MAXS     64                 // upper bound on #valid frames (actual 22)

typedef __bf16 bf16_t;
typedef __attribute__((ext_vector_type(16))) bf16_t v16bf;
typedef __attribute__((ext_vector_type(4)))  bf16_t v4bf;
typedef __attribute__((ext_vector_type(8)))  float  v8f;
typedef __attribute__((ext_vector_type(4)))  float  v4f;

// ---------------------------------------------------------------------------
// Tiled GEMM:  C[M,N] (+)= act( A[M,K] * W[K,N] + bias )
// REQUIREMENTS (guaranteed by the launcher): M%128==0, N%64==0, K%32==0,
// rows 16B-aligned. No bounds checks.
// Block tile 128x64x32, 256 threads = 8 waves; wave owns 32x32 (2x2 WMMA).
// Double-buffered LDS: global loads for tile i+1 are issued before the WMMA
// compute of tile i, so load latency hides behind matrix ops; one barrier
// per K-step. LDS row stride 40 bf16 (80B = 5*16B) keeps ds_load_b128
// alignment while spreading the 16 fragment rows across all bank groups.
// ACT: 0=none, 1=relu, 2=prelu(alpha from device scalar)
// ---------------------------------------------------------------------------
#define BM  128
#define BN  64
#define BK  32
#define BKP 40

template <int ACT, bool ACCUM>
__global__ __launch_bounds__(256) void gemm_wmma_bf16(
    const float* __restrict__ A, int lda,
    const float* __restrict__ W, int ldw,
    const float* __restrict__ bias,
    const float* __restrict__ alpha_p,
    float* __restrict__ C, int ldc,
    int M, int N, int K)
{
    __shared__ bf16_t As[2][BM][BKP];   // [buf][row][k]
    __shared__ bf16_t Bs[2][BN][BKP];   // [buf][col][k]  (transposed at staging)

    const int tid  = threadIdx.x;
    const int lane = tid & 31;
    const int wave = tid >> 5;            // 0..7
    const int wm   = (wave & 3) * 32;     // wave M origin within block tile
    const int wn   = (wave >> 2) * 32;    // wave N origin within block tile
    const int half = lane >> 4;           // 0/1
    const int l16  = lane & 15;

    const int m0 = blockIdx.y * BM;
    const int n0 = blockIdx.x * BN;

    // --- per-thread staging coordinates + streaming pointers (set up once)
    int ar[4], ac[4];
    const float* ap[4];
    #pragma unroll
    for (int p = 0; p < 4; ++p) {
        int v = tid + p * 256;            // 0..1023 over 128x8 float4 grid
        ar[p] = v >> 3;
        ac[p] = (v & 7) * 4;
        ap[p] = A + (size_t)(m0 + ar[p]) * lda + ac[p];
    }
    int wr[2], wc[2];
    const float* wp[2];
    #pragma unroll
    for (int p = 0; p < 2; ++p) {
        int v = tid + p * 256;            // 0..511 over 32x16 float4 grid
        wr[p] = v >> 4;
        wc[p] = (v & 15) * 4;
        wp[p] = W + (size_t)wr[p] * ldw + n0 + wc[p];
    }
    const size_t wstep = (size_t)BK * ldw;

    v8f acc[2][2] = {};
    v4f fa[4], fw[2];

    // issue all global loads for the current tile, advance stream pointers
    auto gload = [&]() {
        #pragma unroll
        for (int p = 0; p < 4; ++p) { fa[p] = *(const v4f*)ap[p]; ap[p] += BK; }
        #pragma unroll
        for (int p = 0; p < 2; ++p) { fw[p] = *(const v4f*)wp[p]; wp[p] += wstep; }
        // W is the stream that misses L2 (weights stream through once);
        // prefetch next tile, one K-step of lead time.
        __builtin_prefetch(wp[0], 0, 1);
    };
    // convert fp32 -> bf16 and store into LDS buffer `buf`
    auto lstore = [&](int buf) {
        #pragma unroll
        for (int p = 0; p < 4; ++p)
            *(v4bf*)&As[buf][ar[p]][ac[p]] = __builtin_convertvector(fa[p], v4bf);
        #pragma unroll
        for (int p = 0; p < 2; ++p)
            #pragma unroll
            for (int j = 0; j < 4; ++j)
                Bs[buf][wc[p] + j][wr[p]] = (bf16_t)fw[p][j];
    };
    // fragment loads (two contiguous 16B chunks per lane) + 4 WMMAs
    auto compute = [&](int buf) {
        v16bf afr[2], bfr[2];
        #pragma unroll
        for (int a = 0; a < 2; ++a) {
            int row = wm + a * 16 + l16;
            #pragma unroll
            for (int j = 0; j < 16; ++j) {
                int kk = ((j & 8) << 1) + half * 8 + (j & 7); // K={0..7,16..23}+8*half
                afr[a][j] = As[buf][row][kk];
            }
        }
        #pragma unroll
        for (int b = 0; b < 2; ++b) {
            int col = wn + b * 16 + l16;
            #pragma unroll
            for (int j = 0; j < 16; ++j)
                bfr[b][j] = Bs[buf][col][((j & 8) << 1) + half * 8 + (j & 7)]; // K={0..7,16..23}+8*half
        }
        #pragma unroll
        for (int a = 0; a < 2; ++a)
            #pragma unroll
            for (int b = 0; b < 2; ++b)
                acc[a][b] = __builtin_amdgcn_wmma_f32_16x16x32_bf16(
                    false, afr[a], false, bfr[b], (short)0, acc[a][b], false, false);
    };

    // --- software-pipelined main loop (double-buffered)
    const int nk = K / BK;
    gload();
    lstore(0);
    __syncthreads();
    for (int it = 1; it < nk; ++it) {
        gload();                 // tile `it` in flight
        compute((it - 1) & 1);   // WMMA on tile `it-1` hides load latency
        lstore(it & 1);          // waits loads, fills other buffer
        __syncthreads();
    }
    compute((nk - 1) & 1);

    // --- epilogue
    float alpha = 0.0f;
    if (ACT == 2 && alpha_p) alpha = alpha_p[0];

    #pragma unroll
    for (int a = 0; a < 2; ++a) {
        #pragma unroll
        for (int b = 0; b < 2; ++b) {
            const int n = n0 + wn + b * 16 + l16;
            const float bv = bias ? bias[n] : 0.0f;
            #pragma unroll
            for (int v = 0; v < 8; ++v) {
                int m = m0 + wm + a * 16 + half * 8 + v;  // C: M = 8*half+vgpr, N = lane&15
                float val = acc[a][b][v] + bv;
                if (ACT == 1) val = val > 0.0f ? val : 0.0f;
                if (ACT == 2) val = val >= 0.0f ? val : alpha * val;
                size_t idx = (size_t)m * ldc + n;
                if (ACCUM) C[idx] += val; else C[idx] = val;
            }
        }
    }
}

// ---------------------------------------------------------------------------
// LayerNorm over D=1024: one row per 256-thread block, one float4 per thread
// ---------------------------------------------------------------------------
__global__ __launch_bounds__(256) void layernorm_k(
    const float* __restrict__ x, const float* __restrict__ g,
    const float* __restrict__ b, float* __restrict__ y)
{
    __shared__ float red[256];
    const int row = blockIdx.x;
    const int tid = threadIdx.x;
    v4f xv = ((const v4f*)(x + (size_t)row * DM))[tid];

    float s = xv[0] + xv[1] + xv[2] + xv[3];
    red[tid] = s; __syncthreads();
    for (int o = 128; o > 0; o >>= 1) {
        if (tid < o) red[tid] += red[tid + o];
        __syncthreads();
    }
    float mean = red[0] * (1.0f / DM);
    __syncthreads();
    float vs = 0.0f;
    #pragma unroll
    for (int j = 0; j < 4; ++j) { float d = xv[j] - mean; vs += d * d; }
    red[tid] = vs; __syncthreads();
    for (int o = 128; o > 0; o >>= 1) {
        if (tid < o) red[tid] += red[tid + o];
        __syncthreads();
    }
    float rstd = rsqrtf(red[0] * (1.0f / DM) + 1e-5f);

    v4f gv = ((const v4f*)g)[tid];
    v4f bv = ((const v4f*)b)[tid];
    v4f yv;
    #pragma unroll
    for (int j = 0; j < 4; ++j) yv[j] = (xv[j] - mean) * rstd * gv[j] + bv[j];
    ((v4f*)(y + (size_t)row * DM))[tid] = yv;
}

// ---------------------------------------------------------------------------
// Attention: one block per (b,t); 16 threads per head; S valid frames.
// score+rel = q . (k[s] + lt[valid[s]-t+127])   (skew identity)
// ---------------------------------------------------------------------------
__global__ __launch_bounds__(256) void attention_k(
    const float* __restrict__ q, const float* __restrict__ k,
    const float* __restrict__ v, const float* __restrict__ lt,
    const int* __restrict__ valid, float* __restrict__ o,
    int S, float scale)
{
    __shared__ float sc[NHEADS][MAXS];
    const int bt = blockIdx.x;
    const int b  = bt >> 7;          // / TLEN
    const int t  = bt & (TLEN - 1);
    const int h  = threadIdx.x >> 4;
    const int l  = threadIdx.x & 15;

    const v4f q4 = *(const v4f*)(q + (size_t)bt * DM + h * DHEAD + l * 4);

    for (int s = 0; s < S; ++s) {
        const int x = valid[s] - t + (TLEN - 1);     // in [0, 2T-2]
        const v4f k4 = *(const v4f*)(k  + (size_t)(b * S + s) * DM + h * DHEAD + l * 4);
        const v4f r4 = *(const v4f*)(lt + (size_t)x * DHEAD + l * 4);
        float p = 0.0f;
        #pragma unroll
        for (int i = 0; i < 4; ++i) p += q4[i] * (k4[i] + r4[i]);
        for (int off = 8; off > 0; off >>= 1) p += __shfl_xor(p, off, 16);
        if (l == 0) sc[h][s] = p * scale;
    }
    __syncthreads();

    float mx = -1e30f;
    for (int s = 0; s < S; ++s) mx = fmaxf(mx, sc[h][s]);
    float den = 0.0f;
    for (int s = 0; s < S; ++s) den += __expf(sc[h][s] - mx);
    v4f o4 = {0.f, 0.f, 0.f, 0.f};
    for (int s = 0; s < S; ++s) {
        float w = __expf(sc[h][s] - mx) / den;
        const v4f v4 = *(const v4f*)(v + (size_t)(b * S + s) * DM + h * DHEAD + l * 4);
        #pragma unroll
        for (int i = 0; i < 4; ++i) o4[i] += w * v4[i];
    }
    *(v4f*)(o + (size_t)bt * DM + h * DHEAD + l * 4) = o4;
}

// ---------------------------------------------------------------------------
// Elementwise helpers (float4 everywhere; all sizes are multiples of 4)
// ---------------------------------------------------------------------------
__global__ void concat_mask_k(const float* __restrict__ x, const float* __restrict__ m,
                              float* __restrict__ xcat)
{
    size_t i = (size_t)blockIdx.x * 256 + threadIdx.x;   // vec4 index
    if (i >= (size_t)NTOK * 128) return;                 // 512 floats = 128 vec4 per row
    size_t bt = i >> 7;
    int qd = (int)(i & 127);
    const v4f* xr = (const v4f*)(x + bt * DMOT);
    const v4f* mr = (const v4f*)(m + bt * DMOT);
    v4f out;
    if (qd < 64) {
        v4f xv = xr[qd], mv = mr[qd];
        #pragma unroll
        for (int j = 0; j < 4; ++j) out[j] = xv[j] * mv[j];
    } else {
        out = mr[qd - 64];
    }
    ((v4f*)xcat)[i] = out;
}

__global__ void kpe_t1_k(const float* __restrict__ w1, const float* __restrict__ b1,
                         const float* __restrict__ a, float* __restrict__ t1)
{
    int i = blockIdx.x * 256 + threadIdx.x;
    if (i >= TLEN * DM) return;
    int t = i >> 10, n = i & (DM - 1);
    float k0 = (float)t - (float)(CTXN - 1);
    float k1 = (float)t - (float)(TLEN - 1);
    float vv = k0 * w1[n] + k1 * w1[DM + n] + b1[n];
    float al = a[0];
    t1[i] = vv >= 0.0f ? vv : al * vv;
}

// 256 rows (row 255 is padding; only x in [0,254] is ever read downstream)
__global__ void rpe_t2_k(const float* __restrict__ w1, const float* __restrict__ b1,
                         const float* __restrict__ a, float* __restrict__ t2)
{
    int i = blockIdx.x * 256 + threadIdx.x;
    if (i >= 2 * TLEN * DM) return;
    int x = i >> 10, n = i & (DM - 1);
    float rd = (float)x - (float)(TLEN - 1);
    float vv = rd * w1[n] + b1[n];
    float al = a[0];
    t2[i] = vv >= 0.0f ? vv : al * vv;
}

__global__ void add_kpe_k(float* __restrict__ h, const float* __restrict__ tab)
{
    size_t i = (size_t)blockIdx.x * 256 + threadIdx.x;   // vec4 index
    if (i >= (size_t)NTOK * (DM / 4)) return;
    v4f hv = ((v4f*)h)[i];
    v4f tv = ((const v4f*)tab)[i % ((size_t)TLEN * (DM / 4))];
    #pragma unroll
    for (int j = 0; j < 4; ++j) hv[j] += tv[j];
    ((v4f*)h)[i] = hv;
}

__global__ void add_vec_k(float* __restrict__ h, const float* __restrict__ o, size_t n4)
{
    size_t i = (size_t)blockIdx.x * 256 + threadIdx.x;
    if (i >= n4) return;
    v4f hv = ((v4f*)h)[i];
    v4f ov = ((const v4f*)o)[i];
    #pragma unroll
    for (int j = 0; j < 4; ++j) hv[j] += ov[j];
    ((v4f*)h)[i] = hv;
}

// one block per kv row; one float4 per thread (1024 floats / 256 threads)
__global__ __launch_bounds__(256) void gather_kv_k(
    const float* __restrict__ h, const int* __restrict__ valid,
    float* __restrict__ kv, int S)
{
    int bs = blockIdx.x;                 // 0 .. B*S-1
    int b = bs / S, s = bs - b * S;
    int t = valid[s];
    const v4f* src = (const v4f*)(h + (((size_t)(b * TLEN + t)) << 10));
    v4f* dst = (v4f*)(kv + ((size_t)bs << 10));
    dst[threadIdx.x] = src[threadIdx.x];
}

// ---------------------------------------------------------------------------
// Host launcher
// ---------------------------------------------------------------------------
extern "C" void kernel_launch(void* const* d_in, const int* in_sizes, int n_in,
                              void* d_out, int out_size, void* d_ws, size_t ws_size,
                              hipStream_t stream)
{
    const float* x       = (const float*)d_in[0];
    const float* mask    = (const float*)d_in[1];
    const int*   valid   = (const int*)  d_in[2];
    const float* enc_w1  = (const float*)d_in[3];
    const float* enc_b1  = (const float*)d_in[4];
    const float* enc_a1  = (const float*)d_in[5];
    const float* enc_w2  = (const float*)d_in[6];
    const float* enc_b2  = (const float*)d_in[7];
    const float* enc_a2  = (const float*)d_in[8];
    const float* kpe_w1  = (const float*)d_in[9];
    const float* kpe_b1  = (const float*)d_in[10];
    const float* kpe_a   = (const float*)d_in[11];
    const float* kpe_w2  = (const float*)d_in[12];
    const float* kpe_b2  = (const float*)d_in[13];
    const float* rpe_w1  = (const float*)d_in[14];
    const float* rpe_b1  = (const float*)d_in[15];
    const float* rpe_a   = (const float*)d_in[16];
    const float* rpe_w2  = (const float*)d_in[17];
    const float* rpe_b2  = (const float*)d_in[18];
    const float* at_ln_g = (const float*)d_in[19];
    const float* at_ln_b = (const float*)d_in[20];
    const float* wq      = (const float*)d_in[21];
    const float* wk      = (const float*)d_in[22];
    const float* wv      = (const float*)d_in[23];
    const float* wo      = (const float*)d_in[24];
    const float* wo_b    = (const float*)d_in[25];
    const float* pf_ln_g = (const float*)d_in[26];
    const float* pf_ln_b = (const float*)d_in[27];
    const float* pf_w1   = (const float*)d_in[28];
    const float* pf_b1   = (const float*)d_in[29];
    const float* pf_w2   = (const float*)d_in[30];
    const float* pf_b2   = (const float*)d_in[31];
    const float* ln_g    = (const float*)d_in[32];
    const float* ln_b    = (const float*)d_in[33];
    const float* dec_w1  = (const float*)d_in[34];
    const float* dec_b1  = (const float*)d_in[35];
    const float* dec_a   = (const float*)d_in[36];
    const float* dec_w2  = (const float*)d_in[37];
    const float* dec_b2  = (const float*)d_in[38];

    const int S = in_sizes[2];            // # valid frames (22)

    // ---- workspace carve-out ----
    float* ws = (float*)d_ws;
    size_t off = 0;
    auto carve = [&](size_t n) { float* p = ws + off; off += n; return p; };
    float* xcat   = carve((size_t)NTOK * 2 * DMOT);
    float* h      = carve((size_t)NTOK * DM);
    float* abuf   = carve((size_t)NTOK * DM);
    float* bbuf   = carve((size_t)NTOK * DM);
    float* qbuf   = carve((size_t)NTOK * DM);
    float* obuf   = carve((size_t)NTOK * DM);
    float* kv     = carve((size_t)BATCH * S * DM);
    float* kbuf   = carve((size_t)BATCH * S * DM);
    float* vbuf   = carve((size_t)BATCH * S * DM);
    float* t1     = carve((size_t)TLEN * DM);
    float* kpetab = carve((size_t)TLEN * DM);
    float* t2     = carve((size_t)2 * TLEN * DM);       // padded to 256 rows
    float* ltbuf  = carve((size_t)2 * TLEN * DHEAD);    // padded to 256 rows
    (void)ws_size; (void)n_in; (void)out_size;

    auto gemm = [&](const float* A, int lda, const float* W, int ldw,
                    const float* bias, const float* alpha,
                    float* C, int ldc, int M, int N, int K, int act, bool accum) {
        dim3 g((N + BN - 1) / BN, (M + BM - 1) / BM), blk(256);
        if (accum) {
            gemm_wmma_bf16<0, true><<<g, blk, 0, stream>>>(A, lda, W, ldw, bias, alpha, C, ldc, M, N, K);
        } else {
            if (act == 0)      gemm_wmma_bf16<0, false><<<g, blk, 0, stream>>>(A, lda, W, ldw, bias, alpha, C, ldc, M, N, K);
            else if (act == 1) gemm_wmma_bf16<1, false><<<g, blk, 0, stream>>>(A, lda, W, ldw, bias, alpha, C, ldc, M, N, K);
            else               gemm_wmma_bf16<2, false><<<g, blk, 0, stream>>>(A, lda, W, ldw, bias, alpha, C, ldc, M, N, K);
        }
    };
    auto blocks1d = [](size_t n) { return (unsigned)((n + 255) / 256); };

    // ---- encoder ----
    concat_mask_k<<<blocks1d((size_t)NTOK * 128), 256, 0, stream>>>(x, mask, xcat);
    gemm(xcat, 2 * DMOT, enc_w1, DM, enc_b1, enc_a1, abuf, DM, NTOK, DM, 2 * DMOT, 2, false);
    gemm(abuf, DM, enc_w2, DM, enc_b2, enc_a2, h, DM, NTOK, DM, DM, 2, false);

    // ---- keyframe positional encoding ----
    kpe_t1_k<<<blocks1d((size_t)TLEN * DM), 256, 0, stream>>>(kpe_w1, kpe_b1, kpe_a, t1);
    gemm(t1, DM, kpe_w2, DM, kpe_b2, nullptr, kpetab, DM, TLEN, DM, DM, 0, false);
    add_kpe_k<<<blocks1d((size_t)NTOK * (DM / 4)), 256, 0, stream>>>(h, kpetab);

    // ---- relative position table lt (256x64, row 255 = pad) ----
    rpe_t2_k<<<blocks1d((size_t)2 * TLEN * DM), 256, 0, stream>>>(rpe_w1, rpe_b1, rpe_a, t2);
    gemm(t2, DM, rpe_w2, DHEAD, rpe_b2, nullptr, ltbuf, DHEAD, 2 * TLEN, DHEAD, DM, 0, false);

    const float scale = 0.125f;  // 1/sqrt(64)
    for (int i = 0; i < NLAYERS; ++i) {
        const size_t MM = (size_t)DM * DM;
        // q = LN(h) @ wq
        layernorm_k<<<NTOK, 256, 0, stream>>>(h, at_ln_g + (size_t)i * DM, at_ln_b + (size_t)i * DM, abuf);
        gemm(abuf, DM, wq + (size_t)i * MM, DM, nullptr, nullptr, qbuf, DM, NTOK, DM, DM, 0, false);
        // kv = h[:, valid] (raw h); k,v projections
        gather_kv_k<<<BATCH * S, 256, 0, stream>>>(h, valid, kv, S);
        gemm(kv, DM, wk + (size_t)i * MM, DM, nullptr, nullptr, kbuf, DM, BATCH * S, DM, DM, 0, false);
        gemm(kv, DM, wv + (size_t)i * MM, DM, nullptr, nullptr, vbuf, DM, BATCH * S, DM, DM, 0, false);
        // attention (rel folded via skew identity)
        attention_k<<<NTOK, 256, 0, stream>>>(qbuf, kbuf, vbuf, ltbuf, valid, obuf, S, scale);
        // h += o @ wo + wo_b ; h += o
        gemm(obuf, DM, wo + (size_t)i * MM, DM, wo_b + (size_t)i * DM, nullptr, h, DM, NTOK, DM, DM, 0, true);
        add_vec_k<<<blocks1d((size_t)NTOK * (DM / 4)), 256, 0, stream>>>(h, obuf, (size_t)NTOK * (DM / 4));
        // FFN, chunked over D_FF in 4 slabs of 1024
        layernorm_k<<<NTOK, 256, 0, stream>>>(h, pf_ln_g + (size_t)i * DM, pf_ln_b + (size_t)i * DM, abuf);
        for (int c = 0; c < DFF / DM; ++c) {
            const float* w1c = pf_w1 + (size_t)i * DM * DFF + (size_t)c * DM;
            const float* b1c = pf_b1 + (size_t)i * DFF + (size_t)c * DM;
            const float* w2c = pf_w2 + (size_t)i * DFF * DM + (size_t)c * DM * DM;
            gemm(abuf, DM, w1c, DFF, b1c, nullptr, bbuf, DM, NTOK, DM, DM, 1, false);
            gemm(bbuf, DM, w2c, DM, (c == 0) ? (pf_b2 + (size_t)i * DM) : nullptr, nullptr,
                 h, DM, NTOK, DM, DM, 0, true);
        }
    }

    // ---- decoder ----
    layernorm_k<<<NTOK, 256, 0, stream>>>(h, ln_g, ln_b, abuf);
    gemm(abuf, DM, dec_w1, DM, dec_b1, dec_a, bbuf, DM, NTOK, DM, DM, 2, false);
    gemm(bbuf, DM, dec_w2, DMOT, dec_b2, nullptr, (float*)d_out, DMOT, NTOK, DMOT, DM, 0, false);

    // second tuple output: batch_mask, copied verbatim
    hipMemcpyAsync((float*)d_out + (size_t)NTOK * DMOT, mask,
                   (size_t)NTOK * DMOT * sizeof(float),
                   hipMemcpyDeviceToDevice, stream);
}
